// BaseBlock_21534966022870
// MI455X (gfx1250) — compile-verified, matrix operand order checked by reading, not run on hardware
//
#include <hip/hip_runtime.h>
#include <hip/hip_bf16.h>

#define B_    8
#define N_    4096
#define CIN_  128
#define COUT_ 256
#define SFPS  1024
#define KNN_  24
#define TCH_  256
#define M_    32
#define H_    4

#define CPG   8            // centroids per workgroup in local kernel
#define ROWS  (CPG*KNN_)   // 192 rows (neighbors) per workgroup
#define ESTR  136          // padded LDS stride for edge (128+8)
#define HSTR  264          // padded LDS stride for 256-wide tiles

static __device__ __forceinline__ float geluf(float x) {
    return 0.5f * x * (1.0f + erff(x * 0.70710678118654752f));
}
#define BN_SCALE 0.99999500003749969f

// ---------------- WMMA helpers (CDNA5 bf16 16x16x32, f32 accum) ----------------
typedef __attribute__((ext_vector_type(16))) __bf16 v16bf;
typedef __attribute__((ext_vector_type(8)))  float  v8f;

union FragU { v16bf v; unsigned u[8]; };

// Load a 16x32 bf16 fragment (A layout; B uses same pattern with n on lanes).
// lane: m = lane&15, kh = lane>>4; VGPR j<4 -> K = kh*8 + 2j ; j>=4 -> K = 16 + kh*8 + 2(j-4)
static __device__ __forceinline__ void load_frag(FragU& f, const __bf16* base, int stride,
                                                 int row, int k0, int lane) {
    int m = lane & 15, kh = lane >> 4;
    const __bf16* p = base + (size_t)(row + m) * stride + k0 + kh * 8;
#pragma unroll
    for (int j = 0; j < 4; ++j) f.u[j]     = *(const unsigned*)(p + 2 * j);
#pragma unroll
    for (int j = 0; j < 4; ++j) f.u[4 + j] = *(const unsigned*)(p + 16 + 2 * j);
}

static __device__ __forceinline__ v8f wmma_bf16(const FragU& a, const FragU& b, v8f c) {
    return __builtin_amdgcn_wmma_f32_16x16x32_bf16(false, a.v, false, b.v, (short)0, c, false, false);
}

// monotone float<->uint map (for LDS atomicMax over signed floats)
static __device__ __forceinline__ unsigned mapf(float f) {
    unsigned u = __float_as_uint(f);
    return (u & 0x80000000u) ? ~u : (u | 0x80000000u);
}
static __device__ __forceinline__ float unmapf(unsigned u) {
    return __uint_as_float((u & 0x80000000u) ? (u & 0x7fffffffu) : ~u);
}

// ---------------- weight f32 -> bf16 ----------------
__global__ void cvt_bf16_kernel(const float* __restrict__ s, __bf16* __restrict__ d, int n) {
    int i = blockIdx.x * 256 + threadIdx.x;
    if (i < n) d[i] = (__bf16)s[i];
}

// ---------------- FPS: one block per batch, wave32 shuffle argmax ----------------
__global__ __launch_bounds__(256) void fps_kernel(const float* __restrict__ xyz,
                                                  const int* __restrict__ far0,
                                                  int* __restrict__ fpsIdx,
                                                  float* __restrict__ newXyz) {
    int b = blockIdx.x, tid = threadIdx.x;
    int wave = tid >> 5, lane = tid & 31;
    __shared__ float cpt[3];
    __shared__ float rv[8];
    __shared__ int   ri[8];
    const float* X = xyz + (size_t)b * N_ * 3;

    float px[16], py[16], pz[16], dmin[16];
#pragma unroll
    for (int j = 0; j < 16; ++j) {
        int p = j * 256 + tid;
        px[j] = X[p * 3 + 0]; py[j] = X[p * 3 + 1]; pz[j] = X[p * 3 + 2];
        dmin[j] = 1e10f;
    }
    if (tid == 0) {
        int far = far0[b];
        fpsIdx[b * SFPS + 0] = far;
        float cx = X[far * 3], cy = X[far * 3 + 1], cz = X[far * 3 + 2];
        cpt[0] = cx; cpt[1] = cy; cpt[2] = cz;
        newXyz[((size_t)b * SFPS) * 3 + 0] = cx;
        newXyz[((size_t)b * SFPS) * 3 + 1] = cy;
        newXyz[((size_t)b * SFPS) * 3 + 2] = cz;
    }
    __syncthreads();

    for (int s = 0; s < SFPS; ++s) {
        float cx = cpt[0], cy = cpt[1], cz = cpt[2];
        float bv = -1.0f; int bi = 0x7fffffff;
#pragma unroll
        for (int j = 0; j < 16; ++j) {
            float dx = px[j] - cx, dy = py[j] - cy, dz = pz[j] - cz;
            float d = __fadd_rn(__fadd_rn(__fmul_rn(dx, dx), __fmul_rn(dy, dy)), __fmul_rn(dz, dz));
            dmin[j] = fminf(dmin[j], d);
            if (dmin[j] > bv) { bv = dmin[j]; bi = j * 256 + tid; }
        }
        // wave32 argmax (tie -> lower point index), no barriers
#pragma unroll
        for (int off = 16; off; off >>= 1) {
            float v2 = __shfl_xor(bv, off, 32);
            int   i2 = __shfl_xor(bi, off, 32);
            if (v2 > bv || (v2 == bv && i2 < bi)) { bv = v2; bi = i2; }
        }
        if (lane == 0) { rv[wave] = bv; ri[wave] = bi; }
        __syncthreads();
        if (tid == 0) {
            float wv = rv[0]; int wi = ri[0];
#pragma unroll
            for (int k = 1; k < 8; ++k) {
                float v2 = rv[k]; int i2 = ri[k];
                if (v2 > wv || (v2 == wv && i2 < wi)) { wv = v2; wi = i2; }
            }
            if (s + 1 < SFPS) {
                fpsIdx[b * SFPS + s + 1] = wi;
                float nx = X[wi * 3], ny = X[wi * 3 + 1], nz = X[wi * 3 + 2];
                cpt[0] = nx; cpt[1] = ny; cpt[2] = nz;
                newXyz[((size_t)b * SFPS + s + 1) * 3 + 0] = nx;
                newXyz[((size_t)b * SFPS + s + 1) * 3 + 1] = ny;
                newXyz[((size_t)b * SFPS + s + 1) * 3 + 2] = nz;
            }
        }
        __syncthreads();
    }
}

// ---------------- KNN: one block per centroid ----------------
__global__ __launch_bounds__(128) void knn_kernel(const float* __restrict__ xyz,
                                                  const float* __restrict__ newXyz,
                                                  int* __restrict__ knnIdx) {
    int bs = blockIdx.x, b = bs >> 10;          // bs = b*SFPS + s
    int tid = threadIdx.x;
    __shared__ float dl[128 * KNN_];
    __shared__ int   il[128 * KNN_];
    __shared__ float rv[128];
    __shared__ int   rpi[128], rtd[128], rwin;

    float cx = newXyz[(size_t)bs * 3 + 0], cy = newXyz[(size_t)bs * 3 + 1], cz = newXyz[(size_t)bs * 3 + 2];
    float c2 = __fadd_rn(__fadd_rn(__fmul_rn(cx, cx), __fmul_rn(cy, cy)), __fmul_rn(cz, cz));
    const float* X = xyz + (size_t)b * N_ * 3;

    float* ld = dl + tid * KNN_;
    int*   li = il + tid * KNN_;
    int cnt = 0;
    for (int p = tid; p < N_; p += 128) {
        float x0 = X[p * 3], y0 = X[p * 3 + 1], z0 = X[p * 3 + 2];
        float p2  = __fadd_rn(__fadd_rn(__fmul_rn(x0, x0), __fmul_rn(y0, y0)), __fmul_rn(z0, z0));
        float dot = __fadd_rn(__fadd_rn(__fmul_rn(cx, x0), __fmul_rn(cy, y0)), __fmul_rn(cz, z0));
        float d   = __fadd_rn(__fadd_rn(__fmul_rn(-2.0f, dot), c2), p2);
        if (cnt < KNN_) {
            int j = cnt - 1;
            while (j >= 0 && ld[j] > d) { ld[j + 1] = ld[j]; li[j + 1] = li[j]; --j; }
            ld[j + 1] = d; li[j + 1] = p; ++cnt;
        } else if (d < ld[KNN_ - 1]) {
            int j = KNN_ - 2;
            while (j >= 0 && ld[j] > d) { ld[j + 1] = ld[j]; li[j + 1] = li[j]; --j; }
            ld[j + 1] = d; li[j + 1] = p;
        }
    }
    __syncthreads();

    int cur = 0;
    for (int j = 0; j < KNN_; ++j) {
        bool ok = (cur < cnt);
        rv[tid]  = ok ? ld[cur] : 3e38f;
        rpi[tid] = ok ? li[cur] : 0x7fffffff;
        rtd[tid] = tid;
        __syncthreads();
        for (int off = 64; off; off >>= 1) {
            if (tid < off) {
                float v2 = rv[tid + off];
                if (v2 < rv[tid] || (v2 == rv[tid] && rpi[tid + off] < rpi[tid])) {
                    rv[tid] = v2; rpi[tid] = rpi[tid + off]; rtd[tid] = rtd[tid + off];
                }
            }
            __syncthreads();
        }
        if (tid == 0) { knnIdx[(size_t)bs * KNN_ + j] = rpi[0]; rwin = rtd[0]; }
        __syncthreads();
        if (tid == rwin) ++cur;
        __syncthreads();
    }
}

// ---------------- Fused Local: gather->edge->3 WMMA GEMMs->maxpool ----------------
// Each wave: loop mt, one A fragment feeds two N-arms (nt = wave, wave+8).
__global__ __launch_bounds__(256) void local_kernel(const float* __restrict__ x,
        const int* __restrict__ fpsIdx, const int* __restrict__ knnIdx,
        const __bf16* __restrict__ Wp, const float* __restrict__ gp, const float* __restrict__ bp,
        const __bf16* __restrict__ W1, const float* __restrict__ g1, const float* __restrict__ b1,
        const __bf16* __restrict__ W2, const float* __restrict__ g2, const float* __restrict__ b2,
        float* __restrict__ fBuf) {
    __shared__ __bf16 sE[ROWS * ESTR];       // edge (bf16), stride 136
    __shared__ __bf16 sH[ROWS * HSTR];       // h after gemm1 (bf16), stride 264
    __shared__ __bf16 sR[ROWS * HSTR];       // r after gemm2 (bf16), stride 264
    __shared__ float  sCent[CPG * CIN_];
    __shared__ unsigned sFU[CPG * COUT_];
    __shared__ int sNid[ROWS];

    int blk = blockIdx.x;
    int b = blk / (SFPS / CPG);
    int s0 = (blk % (SFPS / CPG)) * CPG;
    int tid = threadIdx.x, wave = tid >> 5, lane = tid & 31;
    const float* Xb = x + (size_t)b * CIN_ * N_;    // x[b][c][n]

    if (tid < ROWS) sNid[tid] = knnIdx[((size_t)(b * SFPS + s0)) * KNN_ + tid];
    for (int i = tid; i < CPG * COUT_; i += 256) sFU[i] = 0u;
    for (int e = tid; e < CPG * CIN_; e += 256) {
        int cent = e & (CPG - 1), c = e >> 3;
        sCent[cent * CIN_ + c] = Xb[(size_t)c * N_ + fpsIdx[b * SFPS + s0 + cent]];
    }
    __syncthreads();

    // edge = gathered - center, f32 -> bf16 into LDS
    for (int e = tid; e < ROWS * CIN_; e += 256) {
        int c = e / ROWS, r = e % ROWS;
        float xv = Xb[(size_t)c * N_ + sNid[r]];
        sE[r * ESTR + c] = (__bf16)(xv - sCent[(r / KNN_) * CIN_ + c]);
    }
    __syncthreads();

    int nL = wave * 16 + (lane & 15);            // arm 0 output channel
    int nH = (wave + 8) * 16 + (lane & 15);      // arm 1 output channel
    int mh = lane >> 4;

    // GEMM1: h = gelu(bn(edge @ Wproj^T))   [192x128]x[128x256]
    {
        float gsL = gp[nL] * BN_SCALE, bbL = bp[nL];
        float gsH = gp[nH] * BN_SCALE, bbH = bp[nH];
        for (int mt = 0; mt < 12; ++mt) {
            v8f acc0 = {}, acc1 = {};
            for (int k0 = 0; k0 < CIN_; k0 += 32) {
                FragU a, w0, w1;
                load_frag(a, sE, ESTR, mt * 16, k0, lane);
                load_frag(w0, Wp, CIN_, wave * 16, k0, lane);
                load_frag(w1, Wp, CIN_, (wave + 8) * 16, k0, lane);
                acc0 = wmma_bf16(a, w0, acc0);
                acc1 = wmma_bf16(a, w1, acc1);
            }
#pragma unroll
            for (int r = 0; r < 8; ++r) {
                int m = mt * 16 + r + 8 * mh;
                sH[m * HSTR + nL] = (__bf16)geluf(acc0[r] * gsL + bbL);
                sH[m * HSTR + nH] = (__bf16)geluf(acc1[r] * gsH + bbH);
            }
        }
    }
    __syncthreads();

    // GEMM2: r = gelu(bn(h @ Wl1^T))
    {
        float gsL = g1[nL] * BN_SCALE, bbL = b1[nL];
        float gsH = g1[nH] * BN_SCALE, bbH = b1[nH];
        for (int mt = 0; mt < 12; ++mt) {
            v8f acc0 = {}, acc1 = {};
            for (int k0 = 0; k0 < COUT_; k0 += 32) {
                FragU a, w0, w1;
                load_frag(a, sH, HSTR, mt * 16, k0, lane);
                load_frag(w0, W1, COUT_, wave * 16, k0, lane);
                load_frag(w1, W1, COUT_, (wave + 8) * 16, k0, lane);
                acc0 = wmma_bf16(a, w0, acc0);
                acc1 = wmma_bf16(a, w1, acc1);
            }
#pragma unroll
            for (int r = 0; r < 8; ++r) {
                int m = mt * 16 + r + 8 * mh;
                sR[m * HSTR + nL] = (__bf16)geluf(acc0[r] * gsL + bbL);
                sR[m * HSTR + nH] = (__bf16)geluf(acc1[r] * gsH + bbH);
            }
        }
    }
    __syncthreads();

    // GEMM3: h2 = gelu(bn(r @ Wl2^T) + h); f = max_k h2 (LDS atomic max)
    {
        float gsL = g2[nL] * BN_SCALE, bbL = b2[nL];
        float gsH = g2[nH] * BN_SCALE, bbH = b2[nH];
        for (int mt = 0; mt < 12; ++mt) {
            v8f acc0 = {}, acc1 = {};
            for (int k0 = 0; k0 < COUT_; k0 += 32) {
                FragU a, w0, w1;
                load_frag(a, sR, HSTR, mt * 16, k0, lane);
                load_frag(w0, W2, COUT_, wave * 16, k0, lane);
                load_frag(w1, W2, COUT_, (wave + 8) * 16, k0, lane);
                acc0 = wmma_bf16(a, w0, acc0);
                acc1 = wmma_bf16(a, w1, acc1);
            }
#pragma unroll
            for (int r = 0; r < 8; ++r) {
                int m = mt * 16 + r + 8 * mh;
                int cent = (m / KNN_) * COUT_;
                float v0 = geluf(acc0[r] * gsL + bbL + (float)sH[m * HSTR + nL]);
                atomicMax(&sFU[cent + nL], mapf(v0));
                float v1 = geluf(acc1[r] * gsH + bbH + (float)sH[m * HSTR + nH]);
                atomicMax(&sFU[cent + nH], mapf(v1));
            }
        }
    }
    __syncthreads();

    for (int e = tid; e < CPG * COUT_; e += 256) {
        int cent = e >> 8, o = e & 255;
        fBuf[((size_t)(b * SFPS + s0 + cent)) * COUT_ + o] = unmapf(sFU[e]);
    }
}

// ---------------- Channel GEMM 1: r = gelu(bn(f @ Wc1^T)) -> bf16 ----------------
__global__ __launch_bounds__(128) void chan1_kernel(const float* __restrict__ fBuf,
        const __bf16* __restrict__ Wc, const float* __restrict__ g, const float* __restrict__ bi,
        __bf16* __restrict__ rBuf) {
    __shared__ __bf16 sA[16 * HSTR];
    int mt = blockIdx.x, tid = threadIdx.x, wave = tid >> 5, lane = tid & 31;
    size_t row0 = (size_t)mt * 16;
    for (int e = tid; e < 16 * COUT_; e += 128)
        sA[(e >> 8) * HSTR + (e & 255)] = (__bf16)fBuf[row0 * COUT_ + e];
    __syncthreads();
    v8f acc[4] = {};
    for (int k0 = 0; k0 < COUT_; k0 += 32) {
        FragU a, w[4];
        load_frag(a, sA, HSTR, 0, k0, lane);
#pragma unroll
        for (int t = 0; t < 4; ++t) load_frag(w[t], Wc, COUT_, (wave + 4 * t) * 16, k0, lane);
#pragma unroll
        for (int t = 0; t < 4; ++t) acc[t] = wmma_bf16(a, w[t], acc[t]);
    }
    int mh = lane >> 4;
#pragma unroll
    for (int t = 0; t < 4; ++t) {
        int n = (wave + 4 * t) * 16 + (lane & 15);
        float gs = g[n] * BN_SCALE, bb = bi[n];
#pragma unroll
        for (int r = 0; r < 8; ++r) {
            int m = r + 8 * mh;
            rBuf[(row0 + m) * COUT_ + n] = (__bf16)geluf(acc[t][r] * gs + bb);
        }
    }
}

// ---------------- Channel GEMM 2: fout = gelu(bn(r@Wc2^T)+f); writes x_out^T ----------------
__global__ __launch_bounds__(128) void chan2_kernel(const __bf16* __restrict__ rBuf,
        const __bf16* __restrict__ Wc, const float* __restrict__ g, const float* __restrict__ bi,
        const float* __restrict__ fBuf, float* __restrict__ fOut, float* __restrict__ xOut) {
    int mt = blockIdx.x, tid = threadIdx.x, wave = tid >> 5, lane = tid & 31;
    int row0 = mt * 16;
    v8f acc[4] = {};
    for (int k0 = 0; k0 < COUT_; k0 += 32) {
        FragU a, w[4];
        load_frag(a, rBuf, COUT_, row0, k0, lane);   // A directly from global bf16
#pragma unroll
        for (int t = 0; t < 4; ++t) load_frag(w[t], Wc, COUT_, (wave + 4 * t) * 16, k0, lane);
#pragma unroll
        for (int t = 0; t < 4; ++t) acc[t] = wmma_bf16(a, w[t], acc[t]);
    }
    int mh = lane >> 4;
#pragma unroll
    for (int t = 0; t < 4; ++t) {
        int n = (wave + 4 * t) * 16 + (lane & 15);
        float gs = g[n] * BN_SCALE, bb = bi[n];
#pragma unroll
        for (int r = 0; r < 8; ++r) {
            int gr = row0 + r + 8 * mh;
            float v = geluf(acc[t][r] * gs + bb + fBuf[(size_t)gr * COUT_ + n]);
            fOut[(size_t)gr * COUT_ + n] = v;
            int b = gr >> 10, s = gr & 1023;
            xOut[((size_t)b * COUT_ + n) * SFPS + s] = v;   // x_out[b][o][s]
        }
    }
}

// ---------------- K/V projections (f32 exact) ----------------
__global__ __launch_bounds__(256) void kv_kernel(const float* __restrict__ fOut,
        const float* __restrict__ Wk, const float* __restrict__ Wv,
        float* __restrict__ kk, float* __restrict__ vv) {
    __shared__ float sf[COUT_];
    int row = blockIdx.x, tid = threadIdx.x;
    sf[tid] = fOut[(size_t)row * COUT_ + tid];
    __syncthreads();
    const float* wk = Wk + (size_t)tid * COUT_;
    const float* wv = Wv + (size_t)tid * COUT_;
    float aK = 0.f, aV = 0.f;
    for (int c = 0; c < COUT_; ++c) { aK += sf[c] * wk[c]; aV += sf[c] * wv[c]; }
    kk[(size_t)row * TCH_ + tid] = aK;
    vv[(size_t)row * TCH_ + tid] = aV;
}

// ---------------- Local2Former cross attention: one block per (b,m) ----------------
__global__ __launch_bounds__(256) void l2f_kernel(const float* __restrict__ tIn,
        const float* __restrict__ Wq, const float* __restrict__ kk, const float* __restrict__ vv,
        const float* __restrict__ Wo, float* __restrict__ tOut) {
    int bm = blockIdx.x, b = bm >> 5, m = bm & 31, tid = threadIdx.x;
    __shared__ float st[TCH_], sq[TCH_], sa[SFPS], red[256], sav[TCH_];
    st[tid] = tIn[((size_t)b * M_ + m) * TCH_ + tid];
    __syncthreads();
    { const float* w = Wq + (size_t)tid * TCH_; float a = 0.f;
      for (int c = 0; c < TCH_; ++c) a += st[c] * w[c];
      sq[tid] = a; }
    __syncthreads();
    const float* kb = kk + (size_t)b * SFPS * TCH_;
    float loc[4];
#pragma unroll
    for (int j = 0; j < 4; ++j) {
        const float* kr = kb + (size_t)(tid * 4 + j) * TCH_;
        float a = 0.f;
        for (int d = 0; d < TCH_; ++d) a += sq[d] * kr[d];
        loc[j] = a * 0.0625f;                        // TCH^-0.5
    }
    float mx = fmaxf(fmaxf(loc[0], loc[1]), fmaxf(loc[2], loc[3]));
    red[tid] = mx; __syncthreads();
    for (int off = 128; off; off >>= 1) { if (tid < off) red[tid] = fmaxf(red[tid], red[tid + off]); __syncthreads(); }
    mx = red[0]; __syncthreads();
    float sum = 0.f;
#pragma unroll
    for (int j = 0; j < 4; ++j) { float e = expf(loc[j] - mx); sa[tid * 4 + j] = e; sum += e; }
    red[tid] = sum; __syncthreads();
    for (int off = 128; off; off >>= 1) { if (tid < off) red[tid] += red[tid + off]; __syncthreads(); }
    float inv = 1.0f / red[0]; __syncthreads();
#pragma unroll
    for (int j = 0; j < 4; ++j) sa[tid * 4 + j] *= inv;
    __syncthreads();
    const float* vb = vv + (size_t)b * SFPS * TCH_;
    float av = 0.f;
    for (int s = 0; s < SFPS; ++s) av += sa[s] * vb[(size_t)s * TCH_ + tid];
    sav[tid] = av; __syncthreads();
    const float* w = Wo + (size_t)tid * TCH_;
    float o = 0.f;
    for (int e = 0; e < TCH_; ++e) o += sav[e] * w[e];
    tOut[((size_t)b * M_ + m) * TCH_ + tid] = st[tid] + o;
}

// ---------------- Former encoder layer: one block per batch ----------------
static __device__ __forceinline__ void ln_rows(const float* src, float* dst,
        const float* g, const float* bt, float* smu, float* siv, int tid) {
    if (tid < M_) {
        const float* r = src + tid * TCH_;
        float s = 0.f;
        for (int c = 0; c < TCH_; ++c) s += r[c];
        float mu = s * (1.0f / TCH_);
        float v = 0.f;
        for (int c = 0; c < TCH_; ++c) { float d = r[c] - mu; v += d * d; }
        v *= (1.0f / TCH_);
        smu[tid] = mu; siv[tid] = 1.0f / sqrtf(v + 1e-5f);
    }
    __syncthreads();
    for (int o = tid; o < M_ * TCH_; o += 256) {
        int m = o >> 8, c = o & 255;
        dst[o] = (src[o] - smu[m]) * siv[m] * g[c] + bt[c];
    }
    __syncthreads();
}

__global__ __launch_bounds__(256) void former_kernel(const float* __restrict__ Wqkv,
        const float* __restrict__ Wao, const float* __restrict__ ln1g, const float* __restrict__ ln1b,
        const float* __restrict__ ln2g, const float* __restrict__ ln2b,
        const float* __restrict__ Wf1, const float* __restrict__ Wf2, float* __restrict__ tIO) {
    __shared__ float tL[M_ * TCH_], tn[M_ * TCH_], sqkv[M_ * 3 * TCH_], at[H_ * M_ * M_], oo[M_ * TCH_];
    __shared__ float smu[M_], siv[M_];
    float* g1h = sqkv;  // aliased FF hidden [32*512] (qkv dead by then)
    int b = blockIdx.x, tid = threadIdx.x;
    float* T = tIO + (size_t)b * M_ * TCH_;
    for (int i = tid; i < M_ * TCH_; i += 256) tL[i] = T[i];
    __syncthreads();
    ln_rows(tL, tn, ln1g, ln1b, smu, siv, tid);
    for (int o = tid; o < M_ * 3 * TCH_; o += 256) {
        int m = o / 768, j = o % 768;
        const float* w = Wqkv + (size_t)j * TCH_;
        const float* r = tn + m * TCH_;
        float a = 0.f;
        for (int c = 0; c < TCH_; ++c) a += r[c] * w[c];
        sqkv[m * 768 + j] = a;
    }
    __syncthreads();
    if (tid < H_ * M_) {
        int h = tid >> 5, m = tid & 31;
        float lg[M_]; float mx = -3e38f;
        for (int n = 0; n < M_; ++n) {
            const float* qv = sqkv + m * 768 + h * 64;
            const float* kv = sqkv + n * 768 + 256 + h * 64;
            float a = 0.f;
            for (int d = 0; d < 64; ++d) a += qv[d] * kv[d];
            a *= 0.125f;                               // HD^-0.5
            lg[n] = a; mx = fmaxf(mx, a);
        }
        float sum = 0.f;
        for (int n = 0; n < M_; ++n) { float e = expf(lg[n] - mx); at[(h * M_ + m) * M_ + n] = e; sum += e; }
        float inv = 1.0f / sum;
        for (int n = 0; n < M_; ++n) at[(h * M_ + m) * M_ + n] *= inv;
    }
    __syncthreads();
    for (int o = tid; o < M_ * TCH_; o += 256) {
        int m = o >> 8, d = o & 255, h = d >> 6;
        float a = 0.f;
        for (int n = 0; n < M_; ++n) a += at[(h * M_ + m) * M_ + n] * sqkv[n * 768 + 512 + d];
        oo[m * TCH_ + d] = a;
    }
    __syncthreads();
    for (int o = tid; o < M_ * TCH_; o += 256) {
        int m = o >> 8, d = o & 255;
        const float* w = Wao + (size_t)d * TCH_;
        const float* r = oo + m * TCH_;
        float a = 0.f;
        for (int e = 0; e < TCH_; ++e) a += r[e] * w[e];
        tL[o] += a;
    }
    __syncthreads();
    ln_rows(tL, tn, ln2g, ln2b, smu, siv, tid);
    for (int o = tid; o < M_ * 2 * TCH_; o += 256) {
        int m = o >> 9, j = o & 511;
        const float* w = Wf1 + (size_t)j * TCH_;
        const float* r = tn + m * TCH_;
        float a = 0.f;
        for (int c = 0; c < TCH_; ++c) a += r[c] * w[c];
        g1h[m * 512 + j] = geluf(a);
    }
    __syncthreads();
    for (int o = tid; o < M_ * TCH_; o += 256) {
        int m = o >> 8, d = o & 255;
        const float* w = Wf2 + (size_t)d * 512;
        const float* r = g1h + m * 512;
        float a = 0.f;
        for (int e = 0; e < 512; ++e) a += r[e] * w[e];
        T[o] = tL[o] + a;
    }
}

// ---------------- host ----------------
extern "C" void kernel_launch(void* const* d_in, const int* in_sizes, int n_in,
                              void* d_out, int out_size, void* d_ws, size_t ws_size,
                              hipStream_t stream) {
    const float* xyz  = (const float*)d_in[0];
    const float* x    = (const float*)d_in[1];
    const float* t    = (const float*)d_in[2];
    const int*   far0 = (const int*)d_in[3];
    const float* Wproj=(const float*)d_in[4];  const float* gproj=(const float*)d_in[5];  const float* bproj=(const float*)d_in[6];
    const float* Wl1  =(const float*)d_in[7];  const float* gl1  =(const float*)d_in[8];  const float* bl1  =(const float*)d_in[9];
    const float* Wl2  =(const float*)d_in[10]; const float* gl2  =(const float*)d_in[11]; const float* bl2  =(const float*)d_in[12];
    const float* Wc1  =(const float*)d_in[13]; const float* gc1  =(const float*)d_in[14]; const float* bc1  =(const float*)d_in[15];
    const float* Wc2  =(const float*)d_in[16]; const float* gc2  =(const float*)d_in[17]; const float* bc2  =(const float*)d_in[18];
    const float* Wq   =(const float*)d_in[19]; const float* Wk   =(const float*)d_in[20]; const float* Wv   =(const float*)d_in[21];
    const float* Wo   =(const float*)d_in[22];
    const float* ln1g =(const float*)d_in[23]; const float* ln1b =(const float*)d_in[24];
    const float* Wqkv =(const float*)d_in[25]; const float* Wao  =(const float*)d_in[26];
    const float* ln2g =(const float*)d_in[27]; const float* ln2b =(const float*)d_in[28];
    const float* Wf1  =(const float*)d_in[29]; const float* Wf2  =(const float*)d_in[30];

    float* outXyz = (float*)d_out;                          // [B,S,3]
    float* outX   = outXyz + (size_t)B_ * SFPS * 3;         // [B,COUT,S]
    float* outT   = outX + (size_t)B_ * COUT_ * SFPS;       // [B,M,TCH]

    char* w = (char*)d_ws; size_t off = 0;
    auto alloc = [&](size_t bytes) { void* p = w + off; off = (off + bytes + 255) & ~(size_t)255; return p; };
    int*    fpsIdx = (int*)alloc((size_t)B_ * SFPS * 4);
    int*    knnIdx = (int*)alloc((size_t)B_ * SFPS * KNN_ * 4);
    __bf16* bWp = (__bf16*)alloc((size_t)COUT_ * CIN_ * 2);
    __bf16* bW1 = (__bf16*)alloc((size_t)COUT_ * COUT_ * 2);
    __bf16* bW2 = (__bf16*)alloc((size_t)COUT_ * COUT_ * 2);
    __bf16* bC1 = (__bf16*)alloc((size_t)COUT_ * COUT_ * 2);
    __bf16* bC2 = (__bf16*)alloc((size_t)COUT_ * COUT_ * 2);
    float*  fBuf = (float*)alloc((size_t)B_ * SFPS * COUT_ * 4);
    __bf16* rBuf = (__bf16*)alloc((size_t)B_ * SFPS * COUT_ * 2);
    float*  fOut = (float*)alloc((size_t)B_ * SFPS * COUT_ * 4);
    float*  kkB  = (float*)alloc((size_t)B_ * SFPS * TCH_ * 4);
    float*  vB   = (float*)alloc((size_t)B_ * SFPS * TCH_ * 4);

    cvt_bf16_kernel<<<(COUT_*CIN_ + 255)/256, 256, 0, stream>>>(Wproj, bWp, COUT_*CIN_);
    cvt_bf16_kernel<<<(COUT_*COUT_ + 255)/256, 256, 0, stream>>>(Wl1, bW1, COUT_*COUT_);
    cvt_bf16_kernel<<<(COUT_*COUT_ + 255)/256, 256, 0, stream>>>(Wl2, bW2, COUT_*COUT_);
    cvt_bf16_kernel<<<(COUT_*COUT_ + 255)/256, 256, 0, stream>>>(Wc1, bC1, COUT_*COUT_);
    cvt_bf16_kernel<<<(COUT_*COUT_ + 255)/256, 256, 0, stream>>>(Wc2, bC2, COUT_*COUT_);

    fps_kernel<<<B_, 256, 0, stream>>>(xyz, far0, fpsIdx, outXyz);
    knn_kernel<<<B_ * SFPS, 128, 0, stream>>>(xyz, outXyz, knnIdx);
    local_kernel<<<B_ * SFPS / CPG, 256, 0, stream>>>(x, fpsIdx, knnIdx,
        bWp, gproj, bproj, bW1, gl1, bl1, bW2, gl2, bl2, fBuf);
    chan1_kernel<<<B_ * SFPS / 16, 128, 0, stream>>>(fBuf, bC1, gc1, bc1, rBuf);
    chan2_kernel<<<B_ * SFPS / 16, 128, 0, stream>>>(rBuf, bC2, gc2, bc2, fBuf, fOut, outX);
    kv_kernel<<<B_ * SFPS, 256, 0, stream>>>(fOut, Wk, Wv, kkB, vB);
    l2f_kernel<<<B_ * M_, 256, 0, stream>>>(t, Wq, kkB, vB, Wo, outT);
    former_kernel<<<B_, 256, 0, stream>>>(Wqkv, Wao, ln1g, ln1b, ln2g, ln2b, Wf1, Wf2, outT);
}
